// CausalAttention_35184372088918
// MI455X (gfx1250) — matrix operand order repaired and checked by reading, not compiled, not run on hardware
//
#include <hip/hip_runtime.h>

// ---------------------------------------------------------------------------
// Causal MHA for MI455X (gfx1250): bf16 WMMA everywhere, f32 accumulate.
//   B=2, N=2048, DIM=1024, H=16, DH=64, SCALE = DIM^-0.5 = 1/32
// Compute-bound (~68 GFLOP vs ~µs of HBM traffic at 23.3 TB/s) -> all GEMMs
// go through v_wmma_f32_16x16x32_bf16. GEMM tiles are staged into LDS by the
// Tensor Data Mover (tensor_load_to_lds + s_wait_tensorcnt), with TDM padding
// reproducing the conflict-free pitch-40 LDS layout. K/V (8 MB each in bf16)
// are fully L2-resident (192 MB L2), so attention loads fragments directly
// from global memory.
// ---------------------------------------------------------------------------

typedef __bf16 bf16;
typedef unsigned int u32;
typedef __attribute__((ext_vector_type(16))) bf16  v16bf;
typedef __attribute__((ext_vector_type(8)))  bf16  v8bf;
typedef __attribute__((ext_vector_type(8)))  float v8f;
typedef __attribute__((ext_vector_type(4)))  u32   v4u;
typedef __attribute__((ext_vector_type(8)))  int   v8i;
typedef __attribute__((ext_vector_type(4)))  int   v4i;

constexpr int Bsz  = 2;
constexpr int Nseq = 2048;
constexpr int DIMc = 1024;
constexpr int Hh   = 16;
constexpr int DHh  = 64;
constexpr int Mrows = Bsz * Nseq;          // 4096 token rows
constexpr float kScale = 0.03125f;         // 1024^-0.5
constexpr float kNegInf = -3.0e38f;

static __device__ __forceinline__ v16bf cat8(v8bf lo, v8bf hi) {
  return __builtin_shufflevector(lo, hi, 0,1,2,3,4,5,6,7,8,9,10,11,12,13,14,15);
}

static __device__ __forceinline__ v8f wmma_bf16(v16bf a, v16bf b, v8f c) {
  return __builtin_amdgcn_wmma_f32_16x16x32_bf16(
      /*neg_a=*/false, a, /*neg_b=*/false, b,
      /*c_mod=*/(short)0, c, /*reuse_a=*/false, /*reuse_b=*/false);
}

// ---------------------------------------------------------------------------
// TDM: async 2-D tile load global->LDS. data_size=2B. Pad config: after every
// 16 dwords (one 32-elem row = 64B) insert 4 dwords (16B) of padding -> LDS
// row pitch = 80B = 40 bf16, matching the fragment-read layout below.
// ---------------------------------------------------------------------------
static __device__ __forceinline__ void tdm_load_2d(
    u32 lds_off, unsigned long long gaddr,
    u32 tensor_d0, u32 tensor_d1, unsigned long long stride0,
    u32 tile_d0, u32 tile_d1) {
  v4u g0;
  g0[0] = 1u;                                       // count=1, user descriptor
  g0[1] = lds_off;                                  // lds_addr (bytes)
  g0[2] = (u32)(gaddr & 0xffffffffull);             // global_addr[31:0]
  g0[3] = (u32)((gaddr >> 32) & 0x1ffffffull)       // global_addr[56:32]
        | (2u << 30);                               // type = 2 ("image")
  v8i g1;
  // data_size=1(2B) | pad_enable | pad_interval=3(16dw) | pad_amount=3(4dw)
  g1[0] = (int)(0x00010000u | (1u << 20) | (3u << 22) | (3u << 25));
  g1[1] = (int)((tensor_d0 & 0xffffu) << 16);                     // td0[15:0]
  g1[2] = (int)(((tensor_d0 >> 16) & 0xffffu) | ((tensor_d1 & 0xffffu) << 16));
  g1[3] = (int)(((tensor_d1 >> 16) & 0xffffu) | (tile_d0 << 16)); // tile_dim0
  g1[4] = (int)(tile_d1 & 0xffffu);                 // tile_dim1; tile_dim2=0
  g1[5] = (int)(u32)(stride0 & 0xffffffffull);      // tensor_dim0_stride[31:0]
  g1[6] = (int)(u32)((stride0 >> 32) & 0xffffull);  // stride0[47:32]; stride1=0
  g1[7] = 0;
  v4i z4 = {0, 0, 0, 0};
#if defined(__clang_major__) && __clang_major__ >= 23
  v8i z8 = {0, 0, 0, 0, 0, 0, 0, 0};
  __builtin_amdgcn_tensor_load_to_lds(g0, g1, z4, z4, z8, 0);
#else
  __builtin_amdgcn_tensor_load_to_lds(g0, g1, z4, z4, 0);
#endif
}

// ---------------------------------------------------------------------------
// Kernel 1a: fp32 -> bf16 conversion (activations)
// ---------------------------------------------------------------------------
__global__ __launch_bounds__(256)
void cvt_f32_bf16(const float* __restrict__ src, bf16* __restrict__ dst, int n) {
  int i = blockIdx.x * 256 + threadIdx.x;
  if (i < n) dst[i] = (bf16)src[i];
}

// ---------------------------------------------------------------------------
// Kernel 1b: fp32 -> bf16 transposing convert for weights: WT[n][k] = W[k][n].
// LDS-tiled 32x32 so both global read and write are coalesced. Pre-transposed
// weights make the GEMM B tile the same row-major 128x32 shape as the A tile,
// so the TDM can stage both without any in-kernel transpose.
// ---------------------------------------------------------------------------
__global__ __launch_bounds__(256)
void cvt_transpose_bf16(const float* __restrict__ W, bf16* __restrict__ WT,
                        int K, int Ncols) {
  __shared__ float tile[32][33];
  const int t = threadIdx.x;
  const int r = t >> 5, c = t & 31;           // 8 rows x 32 cols per pass
  const int k0 = blockIdx.y * 32, n0 = blockIdx.x * 32;
  #pragma unroll
  for (int rr = 0; rr < 32; rr += 8)
    tile[r + rr][c] = W[(size_t)(k0 + r + rr) * Ncols + n0 + c];
  __syncthreads();
  #pragma unroll
  for (int rr = 0; rr < 32; rr += 8)
    WT[(size_t)(n0 + r + rr) * K + k0 + c] = (bf16)tile[c][r + rr];
}

// ---------------------------------------------------------------------------
// Kernel 2: bf16 WMMA GEMM  C[Mrows x Ncols] = A[Mrows x 1024] * BT^T
//   (BT is the pre-transposed weight, row-major [Ncols x 1024])
//   mode 0: scatter into q  [B,H,N,DH]                     (Ncols = 1024)
//   mode 1: scatter into k [B,H,N,DH] and v^T [B,H,DH,N]   (Ncols = 2048)
//   mode 2: f32 out[row*DIM+col] + bias[col]               (Ncols = 1024)
// Block 256 thr = 8 waves; tile 128x128, BK=32; TDM double-buffered LDS.
// Wave grid 2x4 -> each wave 64x32 = 4x2 WMMA accumulators.
// ---------------------------------------------------------------------------
__global__ __launch_bounds__(256)
void gemm_wmma_kernel(const bf16* __restrict__ A, const bf16* __restrict__ BT,
                      int Ncols, int mode,
                      bf16* __restrict__ qbuf, bf16* __restrict__ kbuf,
                      bf16* __restrict__ vtbuf,
                      const float* __restrict__ bias, float* __restrict__ outf) {
  // pitch 40 (80B = 20 banks) -> conflict-free fragment reads; TDM pad fills it
  __shared__ __align__(16) bf16 As[2][128][40];   // [row][k]
  __shared__ __align__(16) bf16 Bs[2][128][40];   // [col][k]

  const int tid  = threadIdx.x;
  const int lane = tid & 31, wave = tid >> 5;
  const int wm = wave >> 2, wn = wave & 3;
  const int ln = lane & 15, hl = lane >> 4;
  const int row0 = blockIdx.y * 128, n0 = blockIdx.x * 128;
  constexpr int NK = DIMc / 32;

  v8f acc[4][2];
  #pragma unroll
  for (int mf = 0; mf < 4; ++mf)
    #pragma unroll
    for (int nf = 0; nf < 2; ++nf)
      #pragma unroll
      for (int r = 0; r < 8; ++r) acc[mf][nf][r] = 0.f;

  auto issue = [&](int buf, int k0) {
    tdm_load_2d((u32)(size_t)&As[buf][0][0],
                (unsigned long long)(size_t)A + ((size_t)row0 * DIMc + k0) * 2,
                DIMc, Mrows, DIMc, 32, 128);
    tdm_load_2d((u32)(size_t)&Bs[buf][0][0],
                (unsigned long long)(size_t)BT + ((size_t)n0 * DIMc + k0) * 2,
                DIMc, (u32)Ncols, DIMc, 32, 128);
  };

  if (wave == 0) issue(0, 0);

  for (int kt = 0; kt < NK; ++kt) {
    const int buf = kt & 1;
    if (wave == 0) {
      if (kt + 1 < NK) {
        issue(buf ^ 1, (kt + 1) * 32);          // overwrites buffer last read
        __builtin_amdgcn_s_wait_tensorcnt(2);   // TDM in-order: current pair done
      } else {
        __builtin_amdgcn_s_wait_tensorcnt(0);
      }
    }
    __syncthreads();                            // publish LDS tile to all waves

    v16bf af[4], bfr[2];
    #pragma unroll
    for (int mf = 0; mf < 4; ++mf) {
      // A frag: lanes 0-15 rows, K 0..7 & 16..23; lanes 16-31: 8..15 & 24..31
      const bf16* pa = &As[buf][wm * 64 + mf * 16 + ln][8 * hl];
      af[mf] = cat8(*(const v8bf*)pa, *(const v8bf*)(pa + 16));
    }
    #pragma unroll
    for (int nf = 0; nf < 2; ++nf) {
      // B frag: lanes 0-15 cols, K 0..7 & 16..23; lanes 16-31: 8..15 & 24..31
      const bf16* pb = &Bs[buf][wn * 32 + nf * 16 + ln][8 * hl];
      bfr[nf] = cat8(*(const v8bf*)pb, *(const v8bf*)(pb + 16));
    }
    #pragma unroll
    for (int mf = 0; mf < 4; ++mf)
      #pragma unroll
      for (int nf = 0; nf < 2; ++nf)
        acc[mf][nf] = wmma_bf16(af[mf], bfr[nf], acc[mf][nf]);
    __syncthreads();                            // readers done before next TDM
  }

  // epilogue: C layout VGPR r -> row (r | r+8 per half), lane -> col
  #pragma unroll
  for (int mf = 0; mf < 4; ++mf)
    #pragma unroll
    for (int nf = 0; nf < 2; ++nf)
      #pragma unroll
      for (int r = 0; r < 8; ++r) {
        const int grow = row0 + wm * 64 + mf * 16 + r + 8 * hl;
        const int gcol = n0 + wn * 32 + nf * 16 + ln;
        const float v = acc[mf][nf][r];
        if (mode == 2) {
          outf[(size_t)grow * DIMc + gcol] = v + bias[gcol];
        } else {
          const int bb = grow >> 11, t = grow & (Nseq - 1);
          if (mode == 0) {
            const int hh = gcol >> 6, d = gcol & 63;
            qbuf[(((size_t)(bb * Hh + hh)) * Nseq + t) * DHh + d] = (bf16)v;
          } else {
            const int two = gcol >> 10, rem = gcol & 1023;
            const int hh = rem >> 6, d = rem & 63;
            if (two == 0)
              kbuf[(((size_t)(bb * Hh + hh)) * Nseq + t) * DHh + d] = (bf16)v;
            else  // store V transposed so P*V B-frags are contiguous per lane
              vtbuf[(((size_t)(bb * Hh + hh)) * DHh + d) * Nseq + t] = (bf16)v;
          }
        }
      }
}

// ---------------------------------------------------------------------------
// Kernel 3: causal flash attention, bf16 WMMA, online softmax.
// Block = 128 thr (4 waves); each wave owns 32 query rows (2 M-frags).
// Per 32-key step: 8 WMMAs for S = Q K^T, 8 WMMAs for O += P V.
// K/V fragments loaded straight from global (fully L2-resident).
// ---------------------------------------------------------------------------
__global__ __launch_bounds__(128)
void attn_kernel(const bf16* __restrict__ q, const bf16* __restrict__ k,
                 const bf16* __restrict__ vt, bf16* __restrict__ attn) {
  __shared__ __align__(16) bf16 Pl[4][2][16][40];  // per-wave P relayout patch

  const int tid = threadIdx.x, lane = tid & 31, wave = tid >> 5;
  const int ln = lane & 15, hl = lane >> 4;
  const int bh = blockIdx.y, b = bh / Hh, h = bh % Hh;
  const int qw = blockIdx.x * 128 + wave * 32;  // this wave's first query row

  const bf16* Q = q  + (size_t)bh * Nseq * DHh;
  const bf16* K = k  + (size_t)bh * Nseq * DHh;
  const bf16* V = vt + (size_t)bh * DHh * Nseq;

  // Q A-fragments, resident for the whole kernel: 2 mfrags x 2 dh-chunks
  v16bf aq[2][2];
  #pragma unroll
  for (int mf = 0; mf < 2; ++mf)
    #pragma unroll
    for (int c = 0; c < 2; ++c) {
      const bf16* p = Q + (size_t)(qw + mf * 16 + ln) * DHh + 32 * c + 16 * hl;
      aq[mf][c] = cat8(*(const v8bf*)p, *(const v8bf*)(p + 8));
    }

  v8f accO[2][4];
  float mrun[2][8], lrun[2][8];
  #pragma unroll
  for (int mf = 0; mf < 2; ++mf) {
    #pragma unroll
    for (int nt = 0; nt < 4; ++nt)
      #pragma unroll
      for (int r = 0; r < 8; ++r) accO[mf][nt][r] = 0.f;
    #pragma unroll
    for (int r = 0; r < 8; ++r) { mrun[mf][r] = kNegInf; lrun[mf][r] = 0.f; }
  }

  const int nkt = qw / 32 + 1;  // causal: keys only up to this wave's rows
  for (int kt = 0; kt < nkt; ++kt) {
    const int j0 = kt * 32;
    if (kt + 1 < nkt)  // warm L2->L0 for next K tile
      __builtin_prefetch(K + (size_t)(j0 + 32 + ln) * DHh, 0, 3);

    // ---- S = scale * Q K^T over a 32-key strip (two 16-key subtiles) ----
    v8f s[2][2];
    #pragma unroll
    for (int mf = 0; mf < 2; ++mf)
      #pragma unroll
      for (int st = 0; st < 2; ++st)
        #pragma unroll
        for (int r = 0; r < 8; ++r) s[mf][st][r] = 0.f;

    #pragma unroll
    for (int st = 0; st < 2; ++st)
      #pragma unroll
      for (int c = 0; c < 2; ++c) {
        // B frag of K^T: lane -> key column, 32B contiguous along dh
        const bf16* kp = K + (size_t)(j0 + st * 16 + ln) * DHh + 32 * c + 16 * hl;
        v16bf kf = *(const v16bf*)kp;
        s[0][st] = wmma_bf16(aq[0][c], kf, s[0][st]);
        s[1][st] = wmma_bf16(aq[1][c], kf, s[1][st]);
      }

    // ---- online softmax + P relayout (C layout -> A layout via LDS) ----
    v16bf ap[2];
    #pragma unroll
    for (int mf = 0; mf < 2; ++mf) {
      #pragma unroll
      for (int st = 0; st < 2; ++st)
        #pragma unroll
        for (int r = 0; r < 8; ++r) {
          const int row = qw + mf * 16 + r + 8 * hl;
          const int col = j0 + st * 16 + ln;
          const float val = s[mf][st][r] * kScale;
          s[mf][st][r] = (col > row) ? kNegInf : val;
        }
      #pragma unroll
      for (int r = 0; r < 8; ++r) {
        float t = fmaxf(s[mf][0][r], s[mf][1][r]);
        t = fmaxf(t, __shfl_xor(t, 1, 32));
        t = fmaxf(t, __shfl_xor(t, 2, 32));
        t = fmaxf(t, __shfl_xor(t, 4, 32));
        t = fmaxf(t, __shfl_xor(t, 8, 32));   // stays within 16-lane half
        const float mnew  = fmaxf(mrun[mf][r], t);
        const float alpha = __expf(mrun[mf][r] - mnew);
        mrun[mf][r] = mnew;
        const float p0 = __expf(s[mf][0][r] - mnew);
        const float p1 = __expf(s[mf][1][r] - mnew);
        float rs = p0 + p1;
        rs += __shfl_xor(rs, 1, 32);
        rs += __shfl_xor(rs, 2, 32);
        rs += __shfl_xor(rs, 4, 32);
        rs += __shfl_xor(rs, 8, 32);
        lrun[mf][r] = lrun[mf][r] * alpha + rs;
        #pragma unroll
        for (int nt = 0; nt < 4; ++nt) accO[mf][nt][r] *= alpha;
        Pl[wave][mf][r + 8 * hl][ln]      = (bf16)p0;   // 16x32 P tile
        Pl[wave][mf][r + 8 * hl][16 + ln] = (bf16)p1;
      }
      // re-read P as an A fragment (same-wave DS ops are in-order)
      const bf16* pp = &Pl[wave][mf][ln][16 * hl];
      ap[mf] = cat8(*(const v8bf*)pp, *(const v8bf*)(pp + 8));
    }

    // ---- O += P V : V^T layout gives 32B contiguous B-fragments ----
    #pragma unroll
    for (int nt = 0; nt < 4; ++nt) {
      const bf16* vp = V + (size_t)(nt * 16 + ln) * Nseq + j0 + 16 * hl;
      v16bf vf = *(const v16bf*)vp;
      accO[0][nt] = wmma_bf16(ap[0], vf, accO[0][nt]);
      accO[1][nt] = wmma_bf16(ap[1], vf, accO[1][nt]);
    }
  }

  // normalize and store O as bf16 in token-major [B,N,DIM] for out-projection
  #pragma unroll
  for (int mf = 0; mf < 2; ++mf)
    #pragma unroll
    for (int nt = 0; nt < 4; ++nt)
      #pragma unroll
      for (int r = 0; r < 8; ++r) {
        const int row = qw + mf * 16 + r + 8 * hl;
        const float o = accO[mf][nt][r] / lrun[mf][r];
        attn[((size_t)(b * Nseq + row)) * DIMc + h * DHh + nt * 16 + ln] = (bf16)o;
      }
}

// ---------------------------------------------------------------------------
// Host launch
// ---------------------------------------------------------------------------
extern "C" void kernel_launch(void* const* d_in, const int* in_sizes, int n_in,
                              void* d_out, int out_size, void* d_ws, size_t ws_size,
                              hipStream_t stream) {
  (void)in_sizes; (void)n_in; (void)out_size; (void)ws_size;
  const float* x    = (const float*)d_in[0];
  // d_in[1] = mask (causal, computed analytically in-kernel)
  const float* Wq   = (const float*)d_in[2];
  const float* Wkv  = (const float*)d_in[3];
  const float* Wout = (const float*)d_in[4];
  const float* bias = (const float*)d_in[5];
  float* out = (float*)d_out;

  const size_t XN  = (size_t)Mrows * DIMc;        // 4,194,304
  const size_t WN  = (size_t)DIMc * DIMc;         // 1,048,576
  const size_t WKV = 2 * WN;

  bf16* ws     = (bf16*)d_ws;
  bf16* x_b    = ws;                // [4096,1024]
  bf16* wqT    = x_b    + XN;       // [1024,1024]  (transposed: [out,in])
  bf16* wkvT   = wqT    + WN;       // [2048,1024]  (transposed)
  bf16* woutT  = wkvT   + WKV;      // [1024,1024]  (transposed)
  bf16* q_b    = woutT  + WN;       // [B,H,N,DH]
  bf16* k_b    = q_b    + XN;       // [B,H,N,DH]
  bf16* vt_b   = k_b    + XN;       // [B,H,DH,N]
  bf16* attn_b = vt_b   + XN;       // [B,N,DIM]
  // total: 25,165,824 bf16 = 48 MB of workspace

  cvt_f32_bf16<<<(XN + 255) / 256, 256, 0, stream>>>(x, x_b, (int)XN);
  cvt_transpose_bf16<<<dim3(DIMc / 32, DIMc / 32), 256, 0, stream>>>(
      Wq, wqT, DIMc, DIMc);
  cvt_transpose_bf16<<<dim3(2 * DIMc / 32, DIMc / 32), 256, 0, stream>>>(
      Wkv, wkvT, DIMc, 2 * DIMc);
  cvt_transpose_bf16<<<dim3(DIMc / 32, DIMc / 32), 256, 0, stream>>>(
      Wout, woutT, DIMc, DIMc);

  // Q projection: [4096,1024] x [1024,1024] -> scatter q[B,H,N,DH]
  gemm_wmma_kernel<<<dim3(DIMc / 128, Mrows / 128), 256, 0, stream>>>(
      x_b, wqT, DIMc, 0, q_b, nullptr, nullptr, nullptr, nullptr);
  // KV projection: [4096,1024] x [1024,2048] -> scatter k + v^T
  gemm_wmma_kernel<<<dim3(2 * DIMc / 128, Mrows / 128), 256, 0, stream>>>(
      x_b, wkvT, 2 * DIMc, 1, nullptr, k_b, vt_b, nullptr, nullptr);

  // causal flash attention: 16 row-blocks of 128 per (b,h)
  attn_kernel<<<dim3(Nseq / 128, Bsz * Hh), 128, 0, stream>>>(q_b, k_b, vt_b, attn_b);

  // out projection + bias -> f32 output
  gemm_wmma_kernel<<<dim3(DIMc / 128, Mrows / 128), 256, 0, stream>>>(
      attn_b, woutT, DIMc, 2, nullptr, nullptr, nullptr, bias, out);
}
